// Qwen3VLVisionAttention_13932873908524
// MI455X (gfx1250) — matrix operand order repaired and checked by reading, not compiled, not run on hardware
//
#include <hip/hip_runtime.h>
#include <hip/hip_bf16.h>
#include <math.h>

// ---------------------------------------------------------------------------
// Types for CDNA5 WMMA (gfx1250, wave32)
// ---------------------------------------------------------------------------
typedef _Float16 v16h  __attribute__((ext_vector_type(16)));
typedef float    v8f   __attribute__((ext_vector_type(8)));
typedef _Float16 f16x2 __attribute__((ext_vector_type(2)));
typedef _Float16 f16x4 __attribute__((ext_vector_type(4)));
typedef _Float16 f16x8 __attribute__((ext_vector_type(8)));

constexpr int kE   = 1280;   // embed dim
constexpr int kNH  = 16;     // heads
constexpr int kDH  = 80;     // head dim
constexpr int kDHP = 96;     // head dim padded to multiple of 32 (WMMA K)
constexpr int kT   = 1024;   // sequence length
constexpr int kB   = 8;      // batch
constexpr float kScale = 0.11180339887498949f; // 80^-0.5

// 16-bit A-matrix 16x32 layout (ISA 7.12.2)
__device__ __forceinline__ int a_k_of_r(int r, int lh) {
    int base = (r < 4) ? (2 * r) : (16 + 2 * (r - 4));
    return base + 8 * lh;
}
// 16-bit B-matrix 32x16 layout (pairs along K within a lane's column)
__device__ __forceinline__ int b_k_of_r(int r, int lh) {
    int base = (r < 4) ? (2 * r) : (16 + 2 * (r - 4));
    return base + 8 * lh;
}

__device__ __forceinline__ float redmax16(float v) {
#pragma unroll
    for (int sh = 1; sh < 16; sh <<= 1) v = fmaxf(v, __shfl_xor(v, sh, 32));
    return v;
}
__device__ __forceinline__ float redsum16(float v) {
#pragma unroll
    for (int sh = 1; sh < 16; sh <<= 1) v += __shfl_xor(v, sh, 32);
    return v;
}

// ---------------------------------------------------------------------------
// Kernel 1: QKV GEMM.  C[8192,3840] = X[8192,1280] * W^T + b, scattered into
//   Q  [B,H,T,96] / Kt [B,H,96,T] / V [B,H,T,96]   (all f16)
// 256 thr (8 waves), tile 128x128, BK=32.  Software-pipelined staging:
// next K-tile's global loads issue while current tile's 8 WMMAs run.
// ---------------------------------------------------------------------------
__global__ __launch_bounds__(256) void qkv_gemm_kernel(
    const float* __restrict__ X, const float* __restrict__ W,
    const float* __restrict__ bias,
    _Float16* __restrict__ Qo, _Float16* __restrict__ Kt,
    _Float16* __restrict__ Vo)
{
    __shared__ _Float16 lA[128 * 40];   // padded stride 40 (bank spread)
    __shared__ _Float16 lB[128 * 40];

    const int m0   = blockIdx.x * 128;
    const int n0   = blockIdx.y * 128;
    const int tid  = threadIdx.x;
    const int lane = tid & 31;
    const int wv   = tid >> 5;
    const int wm   = wv >> 1;           // 0..3 : 32-row strip
    const int wn   = wv & 1;            // 0..1 : 64-col strip
    const int lh   = lane >> 4;
    const int l16  = lane & 15;

    v8f acc[2][4] = {};

    float4 ra[4], rb[4];                // register-staged tile chunks
    auto load_tiles = [&](int k0) {
#pragma unroll
        for (int c = 0; c < 4; ++c) {
            int idx = tid + 256 * c;    // 128 rows x 8 float4 per tile
            int row = idx >> 3, col = (idx & 7) << 2;
            ra[c] = *(const float4*)(X + (size_t)(m0 + row) * kE + k0 + col);
            rb[c] = *(const float4*)(W + (size_t)(n0 + row) * kE + k0 + col);
        }
    };
    auto store_tiles = [&]() {
#pragma unroll
        for (int c = 0; c < 4; ++c) {
            int idx = tid + 256 * c;
            int row = idx >> 3, col = (idx & 7) << 2;
            f16x4 ha = { (_Float16)ra[c].x, (_Float16)ra[c].y,
                         (_Float16)ra[c].z, (_Float16)ra[c].w };
            *(f16x4*)(lA + row * 40 + col) = ha;
            f16x4 hb = { (_Float16)rb[c].x, (_Float16)rb[c].y,
                         (_Float16)rb[c].z, (_Float16)rb[c].w };
            *(f16x4*)(lB + row * 40 + col) = hb;
        }
    };

    load_tiles(0);
    for (int k0 = 0; k0 < kE; k0 += 32) {
        store_tiles();
        __syncthreads();
        if (k0 + 32 < kE) load_tiles(k0 + 32);   // overlap with WMMAs below

        v16h af[2], bf[4];
#pragma unroll
        for (int i = 0; i < 2; ++i) {
            int m = wm * 32 + i * 16 + l16;
#pragma unroll
            for (int r = 0; r < 8; ++r) {
                f16x2 p = *(const f16x2*)(lA + m * 40 + a_k_of_r(r, lh));
                af[i][2 * r] = p[0]; af[i][2 * r + 1] = p[1];
            }
        }
#pragma unroll
        for (int j = 0; j < 4; ++j) {
            int n = wn * 64 + j * 16 + l16;
#pragma unroll
            for (int r = 0; r < 8; ++r) {
                f16x2 p = *(const f16x2*)(lB + n * 40 + b_k_of_r(r, lh));
                bf[j][2 * r] = p[0]; bf[j][2 * r + 1] = p[1];
            }
        }
#pragma unroll
        for (int i = 0; i < 2; ++i)
#pragma unroll
            for (int j = 0; j < 4; ++j)
                acc[i][j] = __builtin_amdgcn_wmma_f32_16x16x32_f16(
                    false, af[i], false, bf[j], (short)0, acc[i][j], false, false);
        __syncthreads();
    }

    // ---- epilogue: bias + scatter into Q / Kt / V ----
#pragma unroll
    for (int j = 0; j < 4; ++j) {
        int n   = n0 + wn * 64 + j * 16 + l16;
        float bn = bias[n];
        int sec = n / kE;                 // 0=Q 1=K 2=V
        int nn  = n - sec * kE;
        int hh  = nn / kDH;
        int d   = nn - hh * kDH;
#pragma unroll
        for (int i = 0; i < 2; ++i) {
#pragma unroll
            for (int r = 0; r < 8; ++r) {
                int m = m0 + wm * 32 + i * 16 + r + 8 * lh;
                int b = m >> 10, t = m & (kT - 1);
                _Float16 val = (_Float16)(acc[i][j][r] + bn);
                size_t bh = (size_t)b * kNH + hh;
                if (sec == 0)      Qo[(bh * kT + t) * kDHP + d] = val;
                else if (sec == 1) Kt[(bh * kDHP + d) * kT + t] = val;
                else               Vo[(bh * kT + t) * kDHP + d] = val;
            }
        }
    }
}

// ---------------------------------------------------------------------------
// Kernel 2: flash-style attention.  One block = 4 waves x 16 queries = 64 q.
// Grid: (B*H, T/64).  32-key tiles, online softmax, O accum 16x80 per wave.
// K/V staging pipelined: b64 vector loads for tile n+1 in flight during
// tile n's 11 WMMAs.
// ---------------------------------------------------------------------------
__global__ __launch_bounds__(128) void attn_kernel(
    const _Float16* __restrict__ Q, const _Float16* __restrict__ Kt,
    const _Float16* __restrict__ V, _Float16* __restrict__ Out)
{
    __shared__ _Float16 lK[32 * 100];       // [key][d]  (pairs along d)
    __shared__ _Float16 lV[96 * 36];        // [d][key]  (pairs along key)
    __shared__ _Float16 lP[4 * 16 * 32];    // per-wave P transpose buffer

    const int bh   = blockIdx.x;            // b*16 + h
    const int q0   = blockIdx.y * 64;
    const int tid  = threadIdx.x;
    const int lane = tid & 31;
    const int wv   = tid >> 5;
    const int lh   = lane >> 4;
    const int l16  = lane & 15;

    const _Float16* Qb = Q  + (size_t)bh * kT * kDHP;
    const _Float16* Kb = Kt + (size_t)bh * kDHP * kT;
    const _Float16* Vb = V  + (size_t)bh * kT * kDHP;

    // Q fragments (A-layout): 16 queries x 96 (zero-padded), in registers
    v16h qf[3];
    const int qrow = q0 + wv * 16 + l16;
#pragma unroll
    for (int c = 0; c < 3; ++c)
#pragma unroll
        for (int r = 0; r < 8; ++r) {
            f16x2 p = *(const f16x2*)(Qb + (size_t)qrow * kDHP + c * 32 + a_k_of_r(r, lh));
            qf[c][2 * r] = p[0]; qf[c][2 * r + 1] = p[1];
        }

    // register-staged K/V tile chunks: 768 f16x4 chunks per tile, 6 per thread
    f16x4 rk[6], rv[6];
    auto load_kv = [&](int t0) {
#pragma unroll
        for (int i = 0; i < 6; ++i) {
            int cidx = i * 128 + tid;                 // coalesced across lanes
            int d  = cidx >> 3, tc = (cidx & 7) << 2; // K: 96 rows x 8 chunks
            rk[i] = *(const f16x4*)(Kb + (size_t)d * kT + t0 + tc);
            int t  = cidx / 24, dc = (cidx % 24) << 2; // V: 32 rows x 24 chunks
            rv[i] = *(const f16x4*)(Vb + (size_t)(t0 + t) * kDHP + dc);
        }
    };
    auto store_kv = [&]() {
#pragma unroll
        for (int i = 0; i < 6; ++i) {
            int cidx = i * 128 + tid;
            int d  = cidx >> 3, tc = (cidx & 7) << 2;
#pragma unroll
            for (int j = 0; j < 4; ++j) lK[(tc + j) * 100 + d] = rk[i][j];
            int t  = cidx / 24, dc = (cidx % 24) << 2;
#pragma unroll
            for (int j = 0; j < 4; ++j) lV[(dc + j) * 36 + t] = rv[i][j];
        }
    };

    v8f  o[5] = {};
    float mrow[8], lrow[8];
#pragma unroll
    for (int r = 0; r < 8; ++r) { mrow[r] = -INFINITY; lrow[r] = 0.f; }

    load_kv(0);
    for (int t0 = 0; t0 < kT; t0 += 32) {
        store_kv();
        __syncthreads();
        if (t0 + 32 < kT) load_kv(t0 + 32);   // overlap with WMMAs below

        // ---- S = Q * K^T  (2 key frags x 3 K-steps over Dh=96) ----
        v8f s[2] = {};
#pragma unroll
        for (int j = 0; j < 2; ++j)
#pragma unroll
            for (int c = 0; c < 3; ++c) {
                v16h bfr;
#pragma unroll
                for (int r = 0; r < 8; ++r) {
                    f16x2 p = *(const f16x2*)(lK + (j * 16 + l16) * 100 + c * 32 + b_k_of_r(r, lh));
                    bfr[2 * r] = p[0]; bfr[2 * r + 1] = p[1];
                }
                s[j] = __builtin_amdgcn_wmma_f32_16x16x32_f16(
                    false, qf[c], false, bfr, (short)0, s[j], false, false);
            }

        // ---- online softmax (row r+8*lh lives in VGPR r across 16 lanes) ----
        float alpha[8];
        _Float16* pw = lP + wv * 512;
#pragma unroll
        for (int r = 0; r < 8; ++r) {
            float a0 = s[0][r] * kScale;
            float a1 = s[1][r] * kScale;
            float mx = redmax16(fmaxf(a0, a1));
            float mn = fmaxf(mrow[r], mx);
            alpha[r] = __expf(mrow[r] - mn);
            float p0 = __expf(a0 - mn);
            float p1 = __expf(a1 - mn);
            lrow[r] = lrow[r] * alpha[r] + redsum16(p0 + p1);
            mrow[r] = mn;
            _Float16* pp = pw + (r + 8 * lh) * 32;   // P[m][k] row-major
            pp[l16]      = (_Float16)p0;
            pp[16 + l16] = (_Float16)p1;
        }
        // in-wave LDS store -> load transpose hazard: DS is in-order per wave,
        // but keep the compiler and the DS counter honest.
        asm volatile("s_wait_dscnt 0" ::: "memory");

        // ---- P fragment (A-layout, Kdim = 32 keys) ----
        v16h pf;
#pragma unroll
        for (int r = 0; r < 8; ++r) {
            f16x2 p = *(const f16x2*)(pw + l16 * 32 + a_k_of_r(r, lh));
            pf[2 * r] = p[0]; pf[2 * r + 1] = p[1];
        }

        // ---- O = O*alpha + P*V  (5 d-frags cover d=0..79) ----
#pragma unroll
        for (int nf = 0; nf < 5; ++nf) {
#pragma unroll
            for (int r = 0; r < 8; ++r) o[nf][r] *= alpha[r];
            v16h vf;
#pragma unroll
            for (int r = 0; r < 8; ++r) {
                f16x2 p = *(const f16x2*)(lV + (nf * 16 + l16) * 36 + b_k_of_r(r, lh));
                vf[2 * r] = p[0]; vf[2 * r + 1] = p[1];
            }
            o[nf] = __builtin_amdgcn_wmma_f32_16x16x32_f16(
                false, pf, false, vf, (short)0, o[nf], false, false);
        }
        __syncthreads();
    }

    // ---- normalize and store [B,T,H*80] f16 ----
    const int hh = bh & (kNH - 1);
    const int b  = bh >> 4;
#pragma unroll
    for (int r = 0; r < 8; ++r) {
        float inv = 1.0f / lrow[r];
        int q = q0 + wv * 16 + r + 8 * lh;
#pragma unroll
        for (int nf = 0; nf < 5; ++nf) {
            int d = nf * 16 + l16;
            Out[((size_t)b * kT + q) * kE + hh * kDH + d] = (_Float16)(o[nf][r] * inv);
        }
    }
}

// ---------------------------------------------------------------------------
// Kernel 3: output projection.  out[8192,1280] f32 = A(f16) * W^T + b
// Same pipelined staging as kernel 1.
// ---------------------------------------------------------------------------
__global__ __launch_bounds__(256) void proj_gemm_kernel(
    const _Float16* __restrict__ A, const float* __restrict__ W,
    const float* __restrict__ bias, float* __restrict__ out)
{
    __shared__ _Float16 lA[128 * 40];
    __shared__ _Float16 lB[128 * 40];

    const int m0   = blockIdx.x * 128;
    const int n0   = blockIdx.y * 128;
    const int tid  = threadIdx.x;
    const int lane = tid & 31;
    const int wv   = tid >> 5;
    const int wm   = wv >> 1, wn = wv & 1;
    const int lh   = lane >> 4, l16 = lane & 15;

    v8f acc[2][4] = {};

    f16x8  raA[2];
    float4 rbW[4];
    auto load_tiles = [&](int k0) {
#pragma unroll
        for (int c = 0; c < 2; ++c) {
            int ch = tid + 256 * c;             // 512 chunks of 8 f16
            int row = ch >> 2, col = (ch & 3) << 3;
            raA[c] = *(const f16x8*)(A + (size_t)(m0 + row) * kE + k0 + col);
        }
#pragma unroll
        for (int c = 0; c < 4; ++c) {
            int idx = tid + 256 * c;
            int row = idx >> 3, col = (idx & 7) << 2;
            rbW[c] = *(const float4*)(W + (size_t)(n0 + row) * kE + k0 + col);
        }
    };
    auto store_tiles = [&]() {
#pragma unroll
        for (int c = 0; c < 2; ++c) {
            int ch = tid + 256 * c;
            int row = ch >> 2, col = (ch & 3) << 3;
            *(f16x8*)(lA + row * 40 + col) = raA[c];
        }
#pragma unroll
        for (int c = 0; c < 4; ++c) {
            int idx = tid + 256 * c;
            int row = idx >> 3, col = (idx & 7) << 2;
            f16x4 hb = { (_Float16)rbW[c].x, (_Float16)rbW[c].y,
                         (_Float16)rbW[c].z, (_Float16)rbW[c].w };
            *(f16x4*)(lB + row * 40 + col) = hb;
        }
    };

    load_tiles(0);
    for (int k0 = 0; k0 < kE; k0 += 32) {
        store_tiles();
        __syncthreads();
        if (k0 + 32 < kE) load_tiles(k0 + 32);

        v16h af[2], bf[4];
#pragma unroll
        for (int i = 0; i < 2; ++i) {
            int m = wm * 32 + i * 16 + l16;
#pragma unroll
            for (int r = 0; r < 8; ++r) {
                f16x2 p = *(const f16x2*)(lA + m * 40 + a_k_of_r(r, lh));
                af[i][2 * r] = p[0]; af[i][2 * r + 1] = p[1];
            }
        }
#pragma unroll
        for (int j = 0; j < 4; ++j) {
            int n = wn * 64 + j * 16 + l16;
#pragma unroll
            for (int r = 0; r < 8; ++r) {
                f16x2 p = *(const f16x2*)(lB + n * 40 + b_k_of_r(r, lh));
                bf[j][2 * r] = p[0]; bf[j][2 * r + 1] = p[1];
            }
        }
#pragma unroll
        for (int i = 0; i < 2; ++i)
#pragma unroll
            for (int j = 0; j < 4; ++j)
                acc[i][j] = __builtin_amdgcn_wmma_f32_16x16x32_f16(
                    false, af[i], false, bf[j], (short)0, acc[i][j], false, false);
        __syncthreads();
    }

#pragma unroll
    for (int j = 0; j < 4; ++j) {
        int n = n0 + wn * 64 + j * 16 + l16;
        float bn = bias[n];
#pragma unroll
        for (int i = 0; i < 2; ++i)
#pragma unroll
            for (int r = 0; r < 8; ++r) {
                int m = m0 + wm * 32 + i * 16 + r + 8 * lh;
                out[(size_t)m * kE + n] = acc[i][j][r] + bn;
            }
    }
}

// ---------------------------------------------------------------------------
// Host-side launcher
// ---------------------------------------------------------------------------
extern "C" void kernel_launch(void* const* d_in, const int* in_sizes, int n_in,
                              void* d_out, int out_size, void* d_ws, size_t ws_size,
                              hipStream_t stream) {
    (void)in_sizes; (void)n_in; (void)out_size; (void)ws_size;
    const float* X      = (const float*)d_in[0];   // [8,1024,1280]
    const float* qkv_w  = (const float*)d_in[1];   // [3840,1280]
    const float* qkv_b  = (const float*)d_in[2];   // [3840]
    const float* proj_w = (const float*)d_in[3];   // [1280,1280]
    const float* proj_b = (const float*)d_in[4];   // [1280]
    float* out = (float*)d_out;                    // [8,1024,1280] f32

    const size_t seg = (size_t)kB * kNH * kT * kDHP;   // 12,582,912 f16 elems
    _Float16* Q  = (_Float16*)d_ws;
    _Float16* Kt = Q  + seg;
    _Float16* V  = Kt + seg;
    _Float16* AO = V  + seg;                           // [8192,1280] f16

    // zero Q and Kt so the Dh pad (80..95) contributes nothing to QK^T
    hipMemsetAsync(d_ws, 0, seg * 2 * sizeof(_Float16), stream);

    qkv_gemm_kernel<<<dim3(64, 30), 256, 0, stream>>>(X, qkv_w, qkv_b, Q, Kt, V);
    attn_kernel   <<<dim3(kB * kNH, kT / 64), 128, 0, stream>>>(Q, Kt, V, AO);
    proj_gemm_kernel<<<dim3(64, 10), 256, 0, stream>>>(AO, proj_w, proj_b, out);
}
